// LocalSensetiveAttention_14645838479716
// MI455X (gfx1250) — matrix operand order repaired and checked by reading, not compiled, not run on hardware
//
#include <hip/hip_runtime.h>
#include <hip/hip_bf16.h>
#include <math.h>

// ---------------- problem dims (fixed by reference) ----------------
#define B_ 64
#define T_ 2048
#define E_ 128
#define Q_ 1024
#define M_ 512
#define K_ 31
#define F_ 32

typedef __bf16  v16bf __attribute__((ext_vector_type(16)));
typedef float   v8f   __attribute__((ext_vector_type(8)));

// branchless fast tanh: 1 - 2/(exp(2x)+1); exact saturation at +/-inf
__device__ __forceinline__ float fast_tanh(float x) {
    const float ex = __expf(2.0f * x);
    return 1.0f - __fdividef(2.0f, ex + 1.0f);
}

// ---------------- kernel 0: zero context region of d_out ----------------
__global__ void lsa_zero(float* __restrict__ p, int n) {
    int i = blockIdx.x * 256 + threadIdx.x;
    if (i < n) p[i] = 0.0f;
}

// ---------------- kernel 1: pq[b,e] = query[b,:] . Wq[e,:] + bq[e] ----------------
// tiny (16.8 MFLOP, Wq = 512KB, L2 resident). one block per b, thread = e.
__global__ __launch_bounds__(128) void lsa_pq(const float* __restrict__ query,
                                              const float* __restrict__ Wq,
                                              const float* __restrict__ bq,
                                              float* __restrict__ pq) {
    const int b = blockIdx.x;
    const int e = threadIdx.x;                 // 0..127
    const float4* qr = (const float4*)(query + (size_t)b * Q_);
    const float4* wr = (const float4*)(Wq + (size_t)e * Q_);
    float acc = 0.0f;
    #pragma unroll 4
    for (int i = 0; i < Q_ / 4; ++i) {
        float4 a4 = qr[i];
        float4 w4 = wr[i];
        acc = fmaf(a4.x, w4.x, acc);
        acc = fmaf(a4.y, w4.y, acc);
        acc = fmaf(a4.z, w4.z, acc);
        acc = fmaf(a4.w, w4.w, acc);
    }
    pq[b * E_ + e] = acc + bq[e];
}

// ---------------- kernel 2: all-WMMA energy kernel ----------------
// grid (T/128, B), block 256 = 8 waves; wave w owns t-tile [t0, t0+16).
// Stage 1 (conv1d as GEMM):  loc[16t x 32f] = AlignWin[16t x 32k] * ConvW^T[32k x 32f]
//                            -> 2x v_wmma_f32_16x16x32_bf16 (k=31 column zero padded)
// Stage 2 (layout fixup):    D-tile layout -> A-fragment layout via per-wave LDS
// Stage 3 (projection GEMM): ploc[16t x 128e] = loc[16t x 32f] * Wl^T[32f x 128e]
//                            -> 8x v_wmma_f32_16x16x32_bf16 (unrolled)
// Stage 4: energy[b,t] = sum_e v[e]*tanh(ploc + pq + processed_memory), half-wave shfl reduce
//
// bf16 A/B register layouts used here (contiguous k, same order in A and B):
//  A 16x32: lanes 0-15 (M=lane) hold K=0..15 ; lanes 16-31 hold K=16..31
//  B 32x16: lanes 0-15 (N=lane) hold K=0..15       ; lanes 16-31 hold K=16..31
//  C/D 16x16 f32: VGPR r -> M=r (lanes 0-15) / M=r+8 (lanes 16-31), N=lane%16
__global__ __launch_bounds__(256) void lsa_energy(const float* __restrict__ astate,   // [B,T]
                                                  const float* __restrict__ pm,       // [B,T,E]
                                                  const float* __restrict__ conv_w,   // [F,1,K]
                                                  const float* __restrict__ Wl,       // [E,F]
                                                  const float* __restrict__ vvec,     // [E]
                                                  const float* __restrict__ pq,       // [B,E]
                                                  float* __restrict__ energy) {       // [B,T]
    const int b    = blockIdx.y;
    const int tblk = blockIdx.x * 128;
    const int tid  = threadIdx.x;
    const int wave = tid >> 5;
    const int lane = tid & 31;
    const int t0   = tblk + wave * 16;

    __shared__ float s_align[160];          // astate[b][tblk-15 .. tblk+144], zero padded
    __shared__ float s_loc[8][16][33];      // per-wave loc tile for D->A relayout

    for (int i = tid; i < 160; i += 256) {
        int t = tblk - 15 + i;
        s_align[i] = (t >= 0 && t < T_) ? astate[(size_t)b * T_ + t] : 0.0f;
    }
    __syncthreads();

    const int m    = lane & 15;             // M index (t row) / N index (col)
    const int hi   = lane >> 4;             // half-wave select
    const int ecol = m;

    // ---- Stage 1: conv as WMMA ----
    // A1[m][k] = align[t0+m-15+k]  (sliding windows, broadcast-friendly LDS reads)
    const int astart = (t0 - tblk) + m;     // s_align idx of align[t0+m-15]
    v16bf a1;
    #pragma unroll
    for (int j = 0; j < 16; ++j) {
        const int k = hi * 16 + j;
        a1[j] = (__bf16)s_align[astart + k];
    }
    // B1[k][f] = conv_w[f][k], k=31 column zeroed (conv_w is 4KB, L1/L2 hot)
    v16bf b1lo, b1hi;
    #pragma unroll
    for (int j = 0; j < 16; ++j) {
        const int kk = hi * 16 + j;
        b1lo[j] = (kk < K_) ? (__bf16)conv_w[ecol * K_ + kk]        : (__bf16)0.0f;
        b1hi[j] = (kk < K_) ? (__bf16)conv_w[(16 + ecol) * K_ + kk] : (__bf16)0.0f;
    }
    const v8f zero8 = {};
    v8f dlo = __builtin_amdgcn_wmma_f32_16x16x32_bf16(false, a1, false, b1lo,
                                                      (short)0, zero8, false, false);
    v8f dhi = __builtin_amdgcn_wmma_f32_16x16x32_bf16(false, a1, false, b1hi,
                                                      (short)0, zero8, false, false);

    // ---- Stage 2: D layout -> A layout through per-wave LDS slice ----
    float (*loc)[33] = s_loc[wave];
    #pragma unroll
    for (int r = 0; r < 8; ++r) {
        loc[r + 8 * hi][ecol]      = dlo[r];
        loc[r + 8 * hi][16 + ecol] = dhi[r];
    }
    // same-wave LDS RAW: DS ops are in-order, compiler inserts s_wait_dscnt
    v16bf a2;
    #pragma unroll
    for (int j = 0; j < 16; ++j) {
        const int f = hi * 16 + j;
        a2[j] = (__bf16)loc[m][f];
    }

    // ---- Stage 3+4: projection GEMM, fused epilogue ----
    const int    kb        = hi * 16;       // B-fragment K base
    const int    trow_base = t0 + hi * 8;   // lanes >= 16 hold D rows M=r+8
    const size_t pmb       = ((size_t)b * T_) * E_;

    float partial[8];
    #pragma unroll
    for (int r = 0; r < 8; ++r) partial[r] = 0.0f;

    #pragma unroll
    for (int et = 0; et < 8; ++et) {
        const int e = et * 16 + ecol;
        const float* wrow = Wl + (size_t)e * F_ + kb;   // contiguous in f
        v16bf bb;
        #pragma unroll
        for (int j = 0; j < 16; ++j) bb[j] = (__bf16)wrow[j];

        v8f c = __builtin_amdgcn_wmma_f32_16x16x32_bf16(false, a2, false, bb,
                                                        (short)0, zero8, false, false);
        const float pqv = pq[b * E_ + e];
        const float vv  = vvec[e];
        #pragma unroll
        for (int r = 0; r < 8; ++r) {
            const int t = trow_base + r;
            const float x = c[r] + pqv + pm[pmb + (size_t)t * E_ + e];
            partial[r] += fast_tanh(x) * vv;
        }
    }

    // ---- reduce over e (16 lanes per half-wave) -> energy[b][t] ----
    #pragma unroll
    for (int r = 0; r < 8; ++r) {
        float s = partial[r];
        s += __shfl_xor(s, 1);
        s += __shfl_xor(s, 2);
        s += __shfl_xor(s, 4);
        s += __shfl_xor(s, 8);
        if (ecol == 0) energy[(size_t)b * T_ + trow_base + r] = s;
    }
}

// ---------------- kernel 3: per-batch softmax over T + next_state ----------------
__global__ __launch_bounds__(256) void lsa_softmax(const float* __restrict__ energy,
                                                   const unsigned char* __restrict__ mask,
                                                   const float* __restrict__ astate,
                                                   float* __restrict__ out_align,
                                                   float* __restrict__ out_next) {
    const int b = blockIdx.x;
    const int tid = threadIdx.x;
    __shared__ float red[256];

    float ev[8];
    float mx = -INFINITY;
    #pragma unroll
    for (int i = 0; i < 8; ++i) {
        const int t = tid + i * 256;
        float e = energy[(size_t)b * T_ + t];
        if (mask[(size_t)b * T_ + t]) e = -INFINITY;
        ev[i] = e;
        mx = fmaxf(mx, e);
    }
    red[tid] = mx; __syncthreads();
    for (int s = 128; s > 0; s >>= 1) {
        if (tid < s) red[tid] = fmaxf(red[tid], red[tid + s]);
        __syncthreads();
    }
    mx = red[0]; __syncthreads();

    float sum = 0.0f;
    #pragma unroll
    for (int i = 0; i < 8; ++i) { ev[i] = __expf(ev[i] - mx); sum += ev[i]; }
    red[tid] = sum; __syncthreads();
    for (int s = 128; s > 0; s >>= 1) {
        if (tid < s) red[tid] += red[tid + s];
        __syncthreads();
    }
    const float inv = 1.0f / red[0];

    #pragma unroll
    for (int i = 0; i < 8; ++i) {
        const int t = tid + i * 256;
        const float al = ev[i] * inv;
        out_align[(size_t)b * T_ + t] = al;
        out_next[(size_t)b * T_ + t]  = astate[(size_t)b * T_ + t] + al;
    }
}

// ---------------- kernel 4: context[b,m] = sum_t align[b,t] * memory[b,t,m] ----------------
// dominant 256MB stream. grid (1, 16 t-chunks, B); 128 threads x float4 = all 512 m.
__global__ __launch_bounds__(128) void lsa_context(const float* __restrict__ out_align,
                                                   const float* __restrict__ memory,
                                                   float* __restrict__ ctx) {
    const int b  = blockIdx.z;
    const int tc = blockIdx.y;          // 16 chunks of 128 t
    const int tx = threadIdx.x;         // 128 threads
    __shared__ float s_a[128];

    const int tbase = tc * 128;
    s_a[tx] = out_align[(size_t)b * T_ + tbase + tx];
    __syncthreads();

    const float4* mem = (const float4*)(memory + ((size_t)b * T_ + tbase) * M_);
    float4 acc = make_float4(0.f, 0.f, 0.f, 0.f);
    #pragma unroll 4
    for (int i = 0; i < 128; ++i) {
        const float a = s_a[i];
        const float4 mv = mem[(size_t)i * (M_ / 4) + tx];
        acc.x = fmaf(a, mv.x, acc.x);
        acc.y = fmaf(a, mv.y, acc.y);
        acc.z = fmaf(a, mv.z, acc.z);
        acc.w = fmaf(a, mv.w, acc.w);
    }
    float* cb = ctx + (size_t)b * M_ + tx * 4;
    atomicAdd(cb + 0, acc.x);
    atomicAdd(cb + 1, acc.y);
    atomicAdd(cb + 2, acc.z);
    atomicAdd(cb + 3, acc.w);
}

// ---------------- launcher ----------------
extern "C" void kernel_launch(void* const* d_in, const int* in_sizes, int n_in,
                              void* d_out, int out_size, void* d_ws, size_t ws_size,
                              hipStream_t stream) {
    const float*         query  = (const float*)d_in[0];          // [B,Q]
    const float*         astate = (const float*)d_in[1];          // [B,T]
    const float*         memory = (const float*)d_in[2];          // [B,T,M]
    const float*         pm     = (const float*)d_in[3];          // [B,T,E]
    const unsigned char* mask   = (const unsigned char*)d_in[4];  // [B,T] bool
    const float*         Wq     = (const float*)d_in[5];          // [E,Q]
    const float*         bq     = (const float*)d_in[6];          // [E]
    const float*         conv_w = (const float*)d_in[7];          // [F,1,K]
    const float*         Wl     = (const float*)d_in[8];          // [E,F]
    const float*         vvec   = (const float*)d_in[9];          // [E]

    // output layout: context [B,M] | alignments [B,T] | next_state [B,T]
    float* out_ctx   = (float*)d_out;
    float* out_align = out_ctx + (size_t)B_ * M_;
    float* out_next  = out_align + (size_t)B_ * T_;

    float* ws     = (float*)d_ws;
    float* pq     = ws;                       // [B,E]
    float* energy = ws + (size_t)B_ * E_;     // [B,T]

    lsa_zero<<<dim3((B_ * M_ + 255) / 256), dim3(256), 0, stream>>>(out_ctx, B_ * M_);
    lsa_pq<<<dim3(B_), dim3(128), 0, stream>>>(query, Wq, bq, pq);
    lsa_energy<<<dim3(T_ / 128, B_), dim3(256), 0, stream>>>(astate, pm, conv_w, Wl, vvec, pq, energy);
    lsa_softmax<<<dim3(B_), dim3(256), 0, stream>>>(energy, mask, astate, out_align, out_next);
    lsa_context<<<dim3(1, T_ / 128, B_), dim3(128), 0, stream>>>(out_align, memory, out_ctx);
}
